// GQA_30391188586695
// MI455X (gfx1250) — compile-verified, matrix operand order checked by reading, not run on hardware
//
#include <hip/hip_runtime.h>
#include <hip/hip_bf16.h>

// ---------------------------------------------------------------------------
// GQA forward for MI455X (gfx1250): bf16 WMMA everywhere, intermediates kept
// bf16 in workspace (fits in 192MB L2).  Round 2: TDM (tensor_load_to_lds)
// staging of the bf16 A-tile in the output-projection GEMM.
// ---------------------------------------------------------------------------

#define DIMD   128
#define QH     16
#define KVH    4
#define WIN    512
#define BATCH  4
#define SEQ    2048
#define NQKV   3072          // (QH + 2*KVH) * DIMD

typedef __attribute__((ext_vector_type(4)))  __bf16 v4bf;
typedef __attribute__((ext_vector_type(8)))  __bf16 v8bf;
typedef __attribute__((ext_vector_type(16))) __bf16 v16bf;
typedef __attribute__((ext_vector_type(8)))  float  v8f;
typedef __attribute__((ext_vector_type(4)))  int    s4i;
typedef __attribute__((ext_vector_type(8)))  int    s8i;

__device__ __forceinline__ v8f zero8() {
    v8f z = {0.f,0.f,0.f,0.f,0.f,0.f,0.f,0.f};
    return z;
}

// Load a 16-half WMMA fragment: lane points at (row, k-base) of a row-major
// bf16 tile; halves 0..7 = p[0..7], halves 8..15 = p[16..23].  Matches the
// CDNA5 16-bit A (and transposed-B) VGPR layout.  16-byte aligned loads.
__device__ __forceinline__ v16bf load_frag(const __bf16* p) {
    v8bf lo = *(const v8bf*)(p);
    v8bf hi = *(const v8bf*)(p + 16);
    return __builtin_shufflevector(lo, hi,
        0,1,2,3,4,5,6,7,8,9,10,11,12,13,14,15);
}

__device__ __forceinline__ v8f wmma_bf16(v16bf a, v16bf b, v8f c) {
    return __builtin_amdgcn_wmma_f32_16x16x32_bf16(
        /*neg_a=*/false, a, /*neg_b=*/false, b,
        /*c_mod=*/(short)0, c, /*reuse_a=*/false, /*reuse_b=*/false);
}

// ---------------------------------------------------------------------------
// Kernel 1: QKV projection.  C[8192,3072] = x[8192,128] @ Wqkv[128,3072].
// 128x128 tile per block, 8 waves, K=128 staged whole in LDS as bf16.
// Q/K columns -> qkv buffer [b*s][3072]; V columns -> transposed vT buffer
// [b*KVH + kvh][d][s] so attention P@V B-fragments are direct b128 loads.
// (f32 -> bf16 conversion during staging, so TDM is not applicable here.)
// ---------------------------------------------------------------------------
__global__ __launch_bounds__(256) void qkv_gemm_k(
    const float* __restrict__ x, const float* __restrict__ Wqkv,
    __bf16* __restrict__ qkv, __bf16* __restrict__ vT)
{
    __shared__ __bf16 As[128 * 136];   // x tile, row-major [m][k]
    __shared__ __bf16 Bt[128 * 136];   // W tile, transposed [n][k]

    const int m0 = blockIdx.x * 128;
    const int n0 = blockIdx.y * 128;
    const int t  = threadIdx.x;

    // Stage A: 16384 contiguous floats of x, converted to bf16, 4 at a time.
    for (int i = t; i < 128 * 32; i += 256) {
        int r = i >> 5, c = (i & 31) << 2;
        float4 f = *(const float4*)&x[(size_t)(m0 + r) * DIMD + c];
        v4bf pk = { (__bf16)f.x, (__bf16)f.y, (__bf16)f.z, (__bf16)f.w };
        *(v4bf*)&As[r * 136 + c] = pk;
    }
    // Stage Bt (transposed write): coalesced global read across n.
    for (int i = t; i < 128 * 128; i += 256) {
        int k = i >> 7, n = i & 127;
        Bt[n * 136 + k] = (__bf16)Wqkv[(size_t)k * NQKV + n0 + n];
    }
    __syncthreads();

    const int w    = t >> 5;
    const int lane = t & 31;
    const int r    = lane & 15;
    const int kb   = (lane >> 4) << 3;
    const int hi   = lane >> 4;

    v8f acc[8];
#pragma unroll
    for (int n = 0; n < 8; ++n) acc[n] = zero8();

#pragma unroll
    for (int kk = 0; kk < 4; ++kk) {
        v16bf a = load_frag(&As[(w * 16 + r) * 136 + kk * 32 + kb]);
#pragma unroll
        for (int n = 0; n < 8; ++n) {
            v16bf b = load_frag(&Bt[(n * 16 + r) * 136 + kk * 32 + kb]);
            acc[n] = wmma_bf16(a, b, acc[n]);
        }
    }

    if (n0 < (QH + KVH) * DIMD) {                 // Q and K region
#pragma unroll
        for (int n = 0; n < 8; ++n)
#pragma unroll
            for (int i = 0; i < 8; ++i) {
                int row = m0 + w * 16 + i + hi * 8;
                int col = n0 + n * 16 + (lane & 15);
                qkv[(size_t)row * NQKV + col] = (__bf16)acc[n][i];
            }
    } else {                                       // V region -> transposed
#pragma unroll
        for (int n = 0; n < 8; ++n)
#pragma unroll
            for (int i = 0; i < 8; ++i) {
                int row = m0 + w * 16 + i + hi * 8;
                int col = n0 + n * 16 + (lane & 15) - (QH + KVH) * DIMD;
                int bb  = row >> 11, s = row & (SEQ - 1);
                int kvh = col >> 7,  d = col & (DIMD - 1);
                vT[(((size_t)(bb * KVH + kvh) * DIMD + d) << 11) + s] =
                    (__bf16)acc[n][i];
            }
    }
}

// ---------------------------------------------------------------------------
// Kernel 2: windowed causal attention with bias, flash style.
// One wave (32 threads) per (b, h, 16-query tile).  Q, K, V fragments load
// straight from global (L2-resident bf16); only P round-trips through LDS.
// ---------------------------------------------------------------------------
__global__ __launch_bounds__(32) void attn_k(
    const __bf16* __restrict__ qkv, const __bf16* __restrict__ vT,
    __bf16* __restrict__ att)
{
    __shared__ __bf16 Pls[16 * 40];    // P tile staging, row-major [q][key]

    const int qt = blockIdx.x, h = blockIdx.y, b = blockIdx.z;
    const int qs   = qt * 16;
    const int lane = threadIdx.x;
    const int r    = lane & 15;
    const int hi   = lane >> 4;
    const int kb   = hi << 3;

    const float scale = 0.08838834764831845f;            // 1/sqrt(128)
    const float slope = exp2f(-0.5f * (float)(h + 1));   // exp2(-(h+1)*8/16)
    const int   kvoff = (QH + (h >> 2)) * DIMD;
    const size_t vhead = ((size_t)(b * KVH + (h >> 2)) * DIMD) << 11;

    // Q fragments: 4 K-chunks of 32, straight from global.
    const size_t qrow = (size_t)(b * SEQ + qs + r) * NQKV + h * DIMD;
    v16bf qf[4];
#pragma unroll
    for (int c = 0; c < 4; ++c) qf[c] = load_frag(&qkv[qrow + c * 32 + kb]);

    v8f o[8];
    float mrun[8], lrun[8];
#pragma unroll
    for (int n = 0; n < 8; ++n) o[n] = zero8();
#pragma unroll
    for (int i = 0; i < 8; ++i) { mrun[i] = -1e30f; lrun[i] = 0.f; }

    const int klo  = qs - WIN;
    int k0 = (klo <= 0) ? 0 : (klo & ~31);
    const int kend = qs + 15;

    for (; k0 <= kend; k0 += 32) {
        // ---- scores: 16q x 32k, two N-tiles, K fragments direct from global
        v8f sc0 = zero8(), sc1 = zero8();
        const size_t kr0 = (size_t)(b * SEQ + k0 + r) * NQKV + kvoff;
        const size_t kr1 = kr0 + (size_t)16 * NQKV;
#pragma unroll
        for (int c = 0; c < 4; ++c) {
            sc0 = wmma_bf16(qf[c], load_frag(&qkv[kr0 + c * 32 + kb]), sc0);
            sc1 = wmma_bf16(qf[c], load_frag(&qkv[kr1 + c * 32 + kb]), sc1);
        }

        // ---- online softmax (C-layout: row = i + hi*8, col = lane&15)
        float p0[8], p1[8];
#pragma unroll
        for (int i = 0; i < 8; ++i) {
            const int qg  = qs + i + hi * 8;
            const int kg0 = k0 + r, kg1 = kg0 + 16;
            const int rel0 = qg - kg0, rel1 = qg - kg1;
            const bool v0 = (rel0 >= 0) && (rel0 <= WIN);
            const bool v1 = (rel1 >= 0) && (rel1 <= WIN);
            float s0 = v0 ? sc0[i] * scale + slope * (float)rel0 : -1e30f;
            float s1 = v1 ? sc1[i] * scale + slope * (float)rel1 : -1e30f;

            float tm = fmaxf(s0, s1);
            tm = fmaxf(tm, __shfl_xor(tm, 1, 16));
            tm = fmaxf(tm, __shfl_xor(tm, 2, 16));
            tm = fmaxf(tm, __shfl_xor(tm, 4, 16));
            tm = fmaxf(tm, __shfl_xor(tm, 8, 16));

            const float mnew  = fmaxf(mrun[i], tm);
            const float alpha = __expf(mrun[i] - mnew);
            lrun[i] *= alpha;
#pragma unroll
            for (int n = 0; n < 8; ++n) o[n][i] *= alpha;

            p0[i] = v0 ? __expf(s0 - mnew) : 0.f;
            p1[i] = v1 ? __expf(s1 - mnew) : 0.f;
            float ts = p0[i] + p1[i];
            ts += __shfl_xor(ts, 1, 16);
            ts += __shfl_xor(ts, 2, 16);
            ts += __shfl_xor(ts, 4, 16);
            ts += __shfl_xor(ts, 8, 16);
            lrun[i] += ts;
            mrun[i]  = mnew;
        }

        // ---- redistribute P (C layout -> A layout) through LDS
#pragma unroll
        for (int i = 0; i < 8; ++i) {
            const int M = i + hi * 8;
            Pls[M * 40 + r]      = (__bf16)p0[i];
            Pls[M * 40 + 16 + r] = (__bf16)p1[i];
        }
        __syncthreads();   // single-wave group: DS ordering + wait
        v16bf pf = load_frag(&Pls[r * 40 + kb]);

        // ---- P @ V : B fragments direct from transposed vT buffer
#pragma unroll
        for (int n = 0; n < 8; ++n) {
            const __bf16* vp =
                &vT[vhead + (((size_t)(n * 16 + r)) << 11) + k0 + kb];
            o[n] = wmma_bf16(pf, load_frag(vp), o[n]);
        }
    }

    // ---- normalize and store bf16 attention output [b*s][QH*D]
#pragma unroll
    for (int i = 0; i < 8; ++i) {
        const float inv = 1.0f / fmaxf(lrun[i], 1e-20f);
        const int row = b * SEQ + qs + i + hi * 8;
#pragma unroll
        for (int n = 0; n < 8; ++n)
            att[(size_t)row * (QH * DIMD) + h * DIMD + n * 16 + r] =
                (__bf16)(o[n][i] * inv);
    }
}

// ---------------------------------------------------------------------------
// Kernel 3: output projection.  out[8192,128] = att[8192,2048] @ Wp[2048,128]
// 128-row tile per block (full N=128), 8 waves, K staged 64 at a time.
// The bf16 A-tile (att) is staged by the Tensor Data Mover: one
// tensor_load_to_lds per K-slab, 2-D descriptor with LDS padding so the
// 64-elem (128 B) rows land at a 144 B stride (72 elements, 16 B aligned,
// matching the ds_load_b128 fragment path).
// ---------------------------------------------------------------------------
__global__ __launch_bounds__(256) void proj_gemm_k(
    const __bf16* __restrict__ att, const float* __restrict__ Wp,
    float* __restrict__ out)
{
    __shared__ __bf16 As[128 * 72];   // att tile [m][k], TDM-staged
    __shared__ __bf16 Bt[128 * 72];   // Wp tile transposed [n][k]

    const int m0   = blockIdx.x * 128;
    const int t    = threadIdx.x;
    const int w    = t >> 5;
    const int lane = t & 31;
    const int r    = lane & 15;
    const int kb   = (lane >> 4) << 3;
    const int hi   = lane >> 4;

    // Uniform wave id so the TDM issue is a scalar branch (TDM ignores EXEC:
    // every wave that executes the instruction would launch its own DMA).
    const int wv = __builtin_amdgcn_readfirstlane((int)(threadIdx.x >> 5));
    const unsigned lds_as = (unsigned)(size_t)(&As[0]);   // addr[31:0] = LDS offset

    v8f acc[8];
#pragma unroll
    for (int n = 0; n < 8; ++n) acc[n] = zero8();

    for (int k0 = 0; k0 < QH * DIMD; k0 += 64) {
        // ---- TDM: DMA att[m0..m0+127][k0..k0+63] (bf16) into As with padding
        if (wv == 0) {
            unsigned long long ga =
                (unsigned long long)(size_t)&att[(size_t)m0 * (QH * DIMD) + k0];
            s4i g0;
            g0[0] = 1;                                   // count=1, user D#
            g0[1] = (int)lds_as;                         // lds_addr
            g0[2] = (int)(unsigned)(ga & 0xffffffffull); // global_addr[31:0]
            g0[3] = (int)(((unsigned)(ga >> 32) & 0x1ffffffu) // global_addr[56:32]
                          | 0x80000000u);                // type=2 (bits 127:126)
            s8i g1;
            g1[0] = (1 << 16)      // data_size = 2 bytes
                  | (1 << 20)      // pad_enable
                  | (4 << 22)      // pad_interval: 32 DWORDs (one 128B row)
                  | (3 << 25);     // pad_amount: 4 DWORDs -> 144B LDS stride
            g1[1] = (int)(2048u << 16);      // tensor_dim0[15:0]=2048 (hi half)
            g1[2] = (int)(128u  << 16);      // dim0[31:16]=0 | tensor_dim1=128
            g1[3] = (int)(64u   << 16);      // dim1[31:16]=0 | tile_dim0=64
            g1[4] = 128;                     // tile_dim1=128, tile_dim2=0
            g1[5] = 2048;                    // tensor_dim0_stride[31:0]
            g1[6] = 0;                       // stride0[47:32] | stride1[15:0]
            g1[7] = 0;                       // stride1[47:16]
            asm volatile("tensor_load_to_lds %0, %1"
                         :: "s"(g0), "s"(g1) : "memory");
        }

        // ---- Stage Bt transposed with f32 -> bf16 convert (all waves).
        for (int i = t; i < 64 * 128; i += 256) {
            int k = i >> 7, n = i & 127;
            Bt[n * 72 + k] = (__bf16)Wp[(size_t)(k0 + k) * DIMD + n];
        }
        if (k0 + 64 < QH * DIMD)   // prefetch next W slab into L2
            __builtin_prefetch(&Wp[(size_t)(k0 + 64) * DIMD], 0, 0);

        if (wv == 0) __builtin_amdgcn_s_wait_tensorcnt(0);
        __syncthreads();

#pragma unroll
        for (int kk = 0; kk < 2; ++kk) {
            v16bf a = load_frag(&As[(w * 16 + r) * 72 + kk * 32 + kb]);
#pragma unroll
            for (int n = 0; n < 8; ++n)
                acc[n] = wmma_bf16(a,
                    load_frag(&Bt[(n * 16 + r) * 72 + kk * 32 + kb]), acc[n]);
        }
        __syncthreads();   // all ds reads consumed before next slab's DMA
    }

#pragma unroll
    for (int n = 0; n < 8; ++n)
#pragma unroll
        for (int i = 0; i < 8; ++i)
            out[(size_t)(m0 + w * 16 + i + hi * 8) * DIMD + n * 16 + r] =
                acc[n][i];
}

// ---------------------------------------------------------------------------
extern "C" void kernel_launch(void* const* d_in, const int* in_sizes, int n_in,
                              void* d_out, int out_size, void* d_ws, size_t ws_size,
                              hipStream_t stream)
{
    const float* x     = (const float*)d_in[0];   // (4, 2048, 128)
    const float* Wqkv  = (const float*)d_in[1];   // (128, 3072)
    const float* Wproj = (const float*)d_in[2];   // (2048, 128)
    float*       out   = (float*)d_out;           // (4, 2048, 128)

    char* ws = (char*)d_ws;
    const size_t qkv_bytes = (size_t)BATCH * SEQ * NQKV * 2;           // 48 MB
    const size_t vT_bytes  = (size_t)BATCH * KVH * DIMD * SEQ * 2;     //  8 MB
    __bf16* qkv = (__bf16*)ws;
    __bf16* vT  = (__bf16*)(ws + qkv_bytes);
    __bf16* att = (__bf16*)(ws + qkv_bytes + vT_bytes);                // 32 MB

    // 1) QKV projection: M=8192, N=3072 -> 64 x 24 tiles of 128x128.
    qkv_gemm_k<<<dim3(64, 24), 256, 0, stream>>>(x, Wqkv, qkv, vT);

    // 2) Attention: one wave per (16-query tile, head, batch).
    attn_k<<<dim3(SEQ / 16, QH, BATCH), 32, 0, stream>>>(qkv, vT, att);

    // 3) Output projection: M=8192, N=128 -> 64 blocks.
    proj_gemm_k<<<dim3(64, 1), 256, 0, stream>>>(att, Wproj, out);

    (void)in_sizes; (void)n_in; (void)out_size; (void)ws_size;
}
